// GlobalAttention_47974784696390
// MI455X (gfx1250) — matrix operand order repaired and checked by reading, not compiled, not run on hardware
//
#include <hip/hip_runtime.h>
#include <hip/hip_bf16.h>
#include <math.h>

#define NN 500000
#define GG 5000
#define FF 128
#define UU 128
#define HH 8
#define NTILES (NN / 16)          // 31250
#define WPB 10                    // waves per block in score kernel
#define SCORE_BLOCKS (NTILES / WPB) // 3125, exact

typedef __attribute__((ext_vector_type(16))) __bf16 v16bf;
typedef __attribute__((ext_vector_type(8)))  float  v8f;

// workspace layout (bytes):
//   [0,     4096)  : Mnode  f32[128][8]
//   [4096,  8192)  : Mglob  f32[128][8]
//   [8192,  8256)  : cvec   f32[16]
//   [8448, 12544)  : Bpack  bf16[4][32][16]  (per-lane WMMA B fragments)
//   [16384,176384) : gsc    f32[G][8]

// ---------------- Phase 0: fold weights -------------------------------------
__global__ void k_fold(const float* __restrict__ Wn, const float* __restrict__ bn,
                       const float* __restrict__ Wg, const float* __restrict__ bg,
                       const float* __restrict__ Watt, const float* __restrict__ batt,
                       float* __restrict__ ws_f) {
  float* Mnode = ws_f;
  float* Mglob = ws_f + 1024;
  float* cvec  = ws_f + 2048;
  int t = threadIdx.x;            // 1024 threads
  int k = t >> 3, h = t & 7;
  if (blockIdx.x == 0) {
    float acc = 0.f;
    for (int u = 0; u < UU; ++u) acc += Wn[k * UU + u] * Watt[u * HH + h];
    Mnode[k * HH + h] = acc;
  } else {
    float acc = 0.f;
    for (int u = 0; u < UU; ++u) acc += Wg[k * UU + u] * Watt[(UU + u) * HH + h];
    Mglob[k * HH + h] = acc;
    if (t < HH) {
      float c = batt[t];
      for (int u = 0; u < UU; ++u)
        c += bn[u] * Watt[u * HH + t] + bg[u] * Watt[(UU + u) * HH + t];
      cvec[t] = c;
    }
  }
}

// ---------------- Phase 0b: pack B fragments (bf16, WMMA lane layout) -------
__global__ void k_pack(const float* __restrict__ ws_f, unsigned short* __restrict__ bp) {
  const float* Mnode = ws_f;
  __bf16* Bpack = (__bf16*)bp;
  for (int idx = threadIdx.x; idx < 2048; idx += 256) {
    int c   = idx >> 9;           // K-chunk 0..3
    int rem = idx & 511;
    int L   = rem >> 4;           // lane 0..31
    int i   = rem & 15;           // element within lane fragment
    int n   = L & 15;             // B column
    int K   = c * 32 + ((L < 16) ? 0 : 8) + ((i < 8) ? i : i + 8);
    float v = (n < HH) ? Mnode[K * HH + n] : 0.f;
    Bpack[idx] = (__bf16)v;
  }
}

// ---------------- Phase 1: per-graph global scores --------------------------
__global__ void k_gsc(const float* __restrict__ gf, const float* __restrict__ ws_f,
                      float* __restrict__ gsc) {
  int t = blockIdx.x * blockDim.x + threadIdx.x;
  if (t >= GG * HH) return;
  int g = t >> 3, h = t & 7;
  const float* Mglob = ws_f + 1024;
  const float* row = gf + (size_t)g * FF;
  float acc = 0.f;
  for (int k = 0; k < FF; ++k) acc += row[k] * Mglob[k * HH + h];
  gsc[t] = acc;
}

// ---------------- Phase 2: WMMA node scores ---------------------------------
__global__ void __launch_bounds__(WPB * 32)
k_scores(const float* __restrict__ nf, const int* __restrict__ gidx,
         const unsigned short* __restrict__ bp, const float* __restrict__ gsc,
         const float* __restrict__ cvec, float* __restrict__ out) {
  const int wave = threadIdx.x >> 5;
  const int lane = threadIdx.x & 31;
  const int tile = blockIdx.x * WPB + wave;   // always < NTILES (exact grid)
  const int row0 = tile * 16;

  const __bf16* Bpack = (const __bf16*)bp;
  const float* rowp = nf + (size_t)(row0 + (lane & 15)) * FF;
  const int kb = (lane < 16) ? 0 : 8;

  v8f acc = {};
#pragma unroll
  for (int c = 0; c < 4; ++c) {
    const float4 f0 = *(const float4*)(rowp + c * 32 + kb);
    const float4 f1 = *(const float4*)(rowp + c * 32 + kb + 4);
    const float4 f2 = *(const float4*)(rowp + c * 32 + kb + 16);
    const float4 f3 = *(const float4*)(rowp + c * 32 + kb + 20);
    v16bf a;
    a[0]  = (__bf16)f0.x; a[1]  = (__bf16)f0.y; a[2]  = (__bf16)f0.z; a[3]  = (__bf16)f0.w;
    a[4]  = (__bf16)f1.x; a[5]  = (__bf16)f1.y; a[6]  = (__bf16)f1.z; a[7]  = (__bf16)f1.w;
    a[8]  = (__bf16)f2.x; a[9]  = (__bf16)f2.y; a[10] = (__bf16)f2.z; a[11] = (__bf16)f2.w;
    a[12] = (__bf16)f3.x; a[13] = (__bf16)f3.y; a[14] = (__bf16)f3.z; a[15] = (__bf16)f3.w;
    v16bf b = *(const v16bf*)(Bpack + ((size_t)c * 32 + lane) * 16);
    acc = __builtin_amdgcn_wmma_f32_16x16x32_bf16(false, a, false, b,
                                                  (short)0, acc, false, false);
  }

  const int n = lane & 15;                    // output column (head)
  if (n < HH) {
    const int mbase = (lane < 16) ? 0 : 8;
    const float cv = cvec[n];
#pragma unroll
    for (int r = 0; r < 8; ++r) {
      int node = row0 + mbase + r;
      int g = gidx[node];
      out[(size_t)node * HH + n] = acc[r] + gsc[g * HH + n] + cv;
    }
  }
}

// ---------------- Phase 3: segmented softmax (in place on d_out) ------------
__global__ void k_softmax(const int* __restrict__ gidx, float* __restrict__ out) {
  const int g = blockIdx.x;
  // segment bounds via binary search (graph_indices sorted)
  int lo = 0, hi = NN;
  while (lo < hi) { int mid = (lo + hi) >> 1; if (gidx[mid] < g) lo = mid + 1; else hi = mid; }
  const int start = lo;
  hi = NN;
  while (lo < hi) { int mid = (lo + hi) >> 1; if (gidx[mid] < g + 1) lo = mid + 1; else hi = mid; }
  const int end = lo;

  const int t = threadIdx.x;      // 128 threads: slot 0..15 x head 0..7
  const int h = t & 7, slot = t >> 3;
  __shared__ float red[128];

  float m = -3.402823466e38f;
  for (int i = start + slot; i < end; i += 16)
    m = fmaxf(m, out[(size_t)i * HH + h]);
  red[t] = m; __syncthreads();
  for (int s = 64; s >= 8; s >>= 1) {
    if (t < s) red[t] = fmaxf(red[t], red[t + s]);
    __syncthreads();
  }
  const float mx = red[h];
  __syncthreads();

  float ssum = 0.f;
  for (int i = start + slot; i < end; i += 16)
    ssum += expf(out[(size_t)i * HH + h] - mx);
  red[t] = ssum; __syncthreads();
  for (int s = 64; s >= 8; s >>= 1) {
    if (t < s) red[t] += red[t + s];
    __syncthreads();
  }
  const float denom = red[h];

  for (int i = start + slot; i < end; i += 16)
    out[(size_t)i * HH + h] = expf(out[(size_t)i * HH + h] - mx) / denom;
}

// ---------------- Launcher ---------------------------------------------------
extern "C" void kernel_launch(void* const* d_in, const int* in_sizes, int n_in,
                              void* d_out, int out_size, void* d_ws, size_t ws_size,
                              hipStream_t stream) {
  const float* nf   = (const float*)d_in[0];
  const float* gf   = (const float*)d_in[1];
  const int*   gidx = (const int*)d_in[2];
  const float* Wn   = (const float*)d_in[3];
  const float* bn   = (const float*)d_in[4];
  const float* Wg   = (const float*)d_in[5];
  const float* bg   = (const float*)d_in[6];
  const float* Watt = (const float*)d_in[7];
  const float* batt = (const float*)d_in[8];
  float* out = (float*)d_out;

  float*          ws_f  = (float*)d_ws;
  const float*    cvec  = ws_f + 2048;
  unsigned short* Bpack = (unsigned short*)((char*)d_ws + 8448);
  float*          gsc   = (float*)((char*)d_ws + 16384);

  hipLaunchKernelGGL(k_fold, dim3(2), dim3(1024), 0, stream,
                     Wn, bn, Wg, bg, Watt, batt, ws_f);
  hipLaunchKernelGGL(k_pack, dim3(1), dim3(256), 0, stream, ws_f, Bpack);
  hipLaunchKernelGGL(k_gsc, dim3((GG * HH + 255) / 256), dim3(256), 0, stream,
                     gf, ws_f, gsc);
  hipLaunchKernelGGL(k_scores, dim3(SCORE_BLOCKS), dim3(WPB * 32), 0, stream,
                     nf, gidx, Bpack, gsc, cvec, out);
  hipLaunchKernelGGL(k_softmax, dim3(GG), dim3(128), 0, stream, gidx, out);
}
